// CRF_29497835389233
// MI455X (gfx1250) — matrix operand order repaired and checked by reading, not compiled, not run on hardware
//
#include <hip/hip_runtime.h>

typedef __attribute__((ext_vector_type(16))) _Float16 v16h;
typedef __attribute__((ext_vector_type(8)))  _Float16 v8h;
typedef __attribute__((ext_vector_type(8)))  float    v8f;

#define Bsz 512
#define Tsz 512
#define Lsz 128
#define PV  132   // s_prev row stride in floats  (528 B, 16B-aligned, bank-spread)
#define PA  136   // s_pA   row stride in halves  (272 B, 16B-aligned, bank-spread)

#define LOG2E 1.4426950408889634f   // log2(e)
#define LN2   0.6931471805599453f   // ln(2)

// Raw hardware transcendentals: v_exp_f32 / v_log_f32 are base-2.
// Safe here: exp args are <= 0 (underflow to 0 is correct), log args are
// in (0.5, ~220) (no denorm/inf/nan), so we skip libm's range fixups.
__device__ __forceinline__ float fast_exp(float x) {         // e^x
    return __builtin_amdgcn_exp2f(x * LOG2E);
}
__device__ __forceinline__ float fast_log(float x) {         // ln(x)
    return __builtin_amdgcn_logf(x) * LN2;
}

__global__ __launch_bounds__(256)
void crf_fwd_kernel(const float* __restrict__ feats,
                    const float* __restrict__ transfer,
                    const int*   __restrict__ target,
                    const int*   __restrict__ startp,
                    const int*   __restrict__ stopp,
                    float*       __restrict__ out)
{
    __shared__ float    s_prev[16 * PV];   // running log-alpha, f32, LDS-resident
    __shared__ _Float16 s_pA[16 * PA];     // exp(v - m) in f16, A-operand staging
    __shared__ float    s_m[16];           // per-row max

    const int tid  = threadIdx.x;
    const int wave = tid >> 5;             // 0..7  : owns j-tile [16w, 16w+16)
    const int lane = tid & 31;
    const int b0   = blockIdx.x * 16;      // batch tile base
    const int r    = tid >> 4;             // phase-1 row 0..15
    const int ci   = (tid & 15) * 8;       // phase-1 col base, 8 cols per thread

    const int start = startp[0];
    const int stop  = stopp[0];

    // ---------------- B fragments: E = exp(transfer), constant over the scan --
    // B is K x N (32 x 16) per chunk. Lane n<16: N=n, K pairs (2q,2q+1) q=0..7;
    // lanes 16..31: N=n-16, K pairs 16..31.  (contiguous k order, same as A)
    v16h Bfrag[4];
    {
        const int n   = lane & 15;
        const int col = wave * 16 + n;
        const int kH  = (lane < 16) ? 0 : 16;
#pragma unroll
        for (int c = 0; c < 4; ++c) {
            union { v16h v; _Float16 e[16]; } u;
            const int kb = c * 32 + kH;
#pragma unroll
            for (int q = 0; q < 16; ++q)
                u.e[q] = (_Float16)fast_exp(transfer[(kb + q) * Lsz + col]);
            Bfrag[c] = u.v;
        }
    }

    // ---------------- prev0 = feats[:,1,:] + transfer[start,:] ----------------
    {
        const float* f1 = feats + ((size_t)(b0 + r) * Tsz + 1) * Lsz + ci;
#pragma unroll
        for (int u = 0; u < 8; ++u)
            s_prev[r * PV + ci + u] = f1[u] + transfer[start * Lsz + ci + u];
    }
    __syncthreads();

    // ---------------- scan: t = 2 .. T-1  (510 steps) -------------------------
    for (int t = 2; t < Tsz; ++t) {
        // phase 1: v = prev + feats[:,t]; m = rowmax(v); p = exp(v-m) -> f16 LDS
        const float* ft = feats + ((size_t)(b0 + r) * Tsz + t) * Lsz + ci;
        float v[8];
        float lmax = -3.4e38f;
#pragma unroll
        for (int u = 0; u < 8; ++u) {
            v[u] = s_prev[r * PV + ci + u] + ft[u];
            lmax = fmaxf(lmax, v[u]);
        }
        if (t + 1 < Tsz) __builtin_prefetch(ft + Lsz, 0, 1);   // global_prefetch next slice
#pragma unroll
        for (int d = 1; d < 16; d <<= 1)                        // half-wave max reduce
            lmax = fmaxf(lmax, __shfl_xor(lmax, d, 32));
        if ((tid & 15) == 0) s_m[r] = lmax;
        union { v8h v; _Float16 e[8]; } ph;
#pragma unroll
        for (int u = 0; u < 8; ++u)
            ph.e[u] = (_Float16)__builtin_amdgcn_exp2f((v[u] - lmax) * LOG2E);
        *(v8h*)&s_pA[r * PA + ci] = ph.v;                       // 16B LDS store
        __syncthreads();

        // phase 2: each wave computes its 16x16 tile of p @ E, K=128 in 4 WMMAs
        v8f acc = {0.f, 0.f, 0.f, 0.f, 0.f, 0.f, 0.f, 0.f};
#pragma unroll
        for (int c = 0; c < 4; ++c) {
            // A layout: lane&15 = M; lanes 0..15 hold K {0..15},
            // lanes 16..31 hold K {16..31} within the 32-wide chunk (matches Bfrag).
            const _Float16* rp =
                &s_pA[(lane & 15) * PA + c * 32 + ((lane < 16) ? 0 : 16)];
            v8h lo = *(const v8h*)(rp);        // K = base+dK .. +7
            v8h hi = *(const v8h*)(rp + 8);    // K = base+dK+8 .. +15
            v16h a;
#pragma unroll
            for (int q = 0; q < 8; ++q) { a[q] = lo[q]; a[q + 8] = hi[q]; }
            acc = __builtin_amdgcn_wmma_f32_16x16x32_f16(
                false, a, false, Bfrag[c], (short)0, acc, false, false);
        }
        // D layout: lanes 0-15 -> M = q, lanes 16-31 -> M = 8+q; N = lane&15.
        // new prev = m + ln(acc); acc in (0.5, ~220) -> raw v_log_f32 * ln2.
        const int rowb = (lane < 16) ? 0 : 8;
        const int colw = wave * 16 + (lane & 15);
#pragma unroll
        for (int q = 0; q < 8; ++q)
            s_prev[(rowb + q) * PV + colw] =
                s_m[rowb + q] + __builtin_amdgcn_logf(acc[q]) * LN2;
        __syncthreads();
    }

    // ---------------- sentence = logsumexp(prev + transfer[:,stop]) -----------
    float sv[8];
    float mx = -3.4e38f;
#pragma unroll
    for (int u = 0; u < 8; ++u) {
        sv[u] = s_prev[r * PV + ci + u] + transfer[(ci + u) * Lsz + stop];
        mx = fmaxf(mx, sv[u]);
    }
#pragma unroll
    for (int d = 1; d < 16; d <<= 1) mx = fmaxf(mx, __shfl_xor(mx, d, 32));
    float ssum = 0.f;
#pragma unroll
    for (int u = 0; u < 8; ++u) ssum += fast_exp(sv[u] - mx);
#pragma unroll
    for (int d = 1; d < 16; d <<= 1) ssum += __shfl_xor(ssum, d, 32);
    const float sentence = mx + fast_log(ssum);

    // ---------------- gold score (gather-sum over t) ---------------------------
    float gsum = 0.f;
    const int brow = b0 + r;
    for (int t = 1 + (tid & 15); t < Tsz; t += 16) {
        const int tg = target[brow * Tsz + t];
        gsum += feats[((size_t)brow * Tsz + t) * Lsz + tg];
        const int pre = (t == 1) ? start : target[brow * Tsz + t - 1];
        gsum += transfer[pre * Lsz + tg];
    }
#pragma unroll
    for (int d = 1; d < 16; d <<= 1) gsum += __shfl_xor(gsum, d, 32);
    if ((tid & 15) == 0) {
        const float emit0 = feats[((size_t)brow * Tsz) * Lsz + start];
        out[brow] = sentence - __expf(emit0 + gsum);
    }
}

extern "C" void kernel_launch(void* const* d_in, const int* in_sizes, int n_in,
                              void* d_out, int out_size, void* d_ws, size_t ws_size,
                              hipStream_t stream)
{
    const float* feats    = (const float*)d_in[0];
    const float* transfer = (const float*)d_in[1];
    const int*   target   = (const int*)d_in[2];
    const int*   startp   = (const int*)d_in[3];
    const int*   stopp    = (const int*)d_in[4];
    float*       out      = (float*)d_out;

    crf_fwd_kernel<<<Bsz / 16, 256, 0, stream>>>(feats, transfer, target,
                                                 startp, stopp, out);
}
